// DKFSTLSTMCell_68212670595230
// MI455X (gfx1250) — compile-verified, matrix operand order checked by reading, not run on hardware
//
#include <hip/hip_runtime.h>
#include <math.h>

// ---------------------------------------------------------------------------
// Types for CDNA5 WMMA (wave32, 16x16x32 bf16 -> f32)
// ---------------------------------------------------------------------------
typedef __attribute__((ext_vector_type(16))) __bf16 bf16x16;
typedef __attribute__((ext_vector_type(8)))  float  v8f;

__device__ __forceinline__ v8f wmma_bf16(bf16x16 a, bf16x16 b, v8f c) {
    // (neg_a, A, neg_b, B, c_mod, C, reuse_a, reuse_b)
    return __builtin_amdgcn_wmma_f32_16x16x32_bf16(false, a, false, b, (short)0, c,
                                                   false, false);
}

__device__ __forceinline__ __bf16 f2bf(float f) {
    unsigned u = __builtin_bit_cast(unsigned, f);
    unsigned r = (u + 0x7FFFu + ((u >> 16) & 1u)) >> 16;
    unsigned short s = (unsigned short)r;
    return __builtin_bit_cast(__bf16, s);
}

__device__ __forceinline__ float sigf(float x) { return 1.0f / (1.0f + __expf(-x)); }
__device__ __forceinline__ float lrelu(float x) { return x > 0.0f ? x : 0.01f * x; }

// Problem constants
#define BB  32
#define HC  64
#define SD  64
#define HH  64
#define WW  64
#define NPIX 4096           // 64*64
#define IMG (HC * NPIX)     // one [HC,64,64] image

// ---------------------------------------------------------------------------
// Packing kernels: build bf16, channels-innermost tensors for WMMA
// ---------------------------------------------------------------------------

// X192[b][y][x][192] : ch 0..63 = 1x1 input conv of x, 64..127 = h, 128..191 = m
__global__ void k_pack_x192(const float* __restrict__ x, const float* __restrict__ h,
                            const float* __restrict__ m, const float* __restrict__ w_in,
                            const float* __restrict__ b_in, __bf16* __restrict__ X192) {
    int idx = blockIdx.x * blockDim.x + threadIdx.x;
    if (idx >= BB * NPIX * 192) return;
    int ci  = idx % 192;
    int pg  = idx / 192;            // global pixel index
    int b   = pg / NPIX;
    int p   = pg % NPIX;
    float v;
    if (ci < 64) {
        float acc = b_in[ci];
        #pragma unroll
        for (int c = 0; c < 3; ++c)
            acc += w_in[ci * 3 + c] * x[(b * 3 + c) * NPIX + p];
        v = acc;
    } else if (ci < 128) {
        v = h[(b * HC + (ci - 64)) * NPIX + p];
    } else {
        v = m[(b * HC + (ci - 128)) * NPIX + p];
    }
    X192[idx] = f2bf(v);
}

// [CO][CI][3][3] f32 -> [9][CO][CI] bf16
__global__ void k_pack_taps(const float* __restrict__ w, __bf16* __restrict__ out,
                            int CO, int CI) {
    int idx = blockIdx.x * blockDim.x + threadIdx.x;
    int total = CO * CI * 9;
    if (idx >= total) return;
    int ci = idx % CI;
    int t  = idx / CI;
    int co = t % CO;
    t /= CO;
    out[idx] = f2bf(w[(co * CI + ci) * 9 + t]);
}

__global__ void k_f32_to_bf16(const float* __restrict__ in, __bf16* __restrict__ out, int n) {
    int i = blockIdx.x * blockDim.x + threadIdx.x;
    if (i < n) out[i] = f2bf(in[i]);
}

// ---------------------------------------------------------------------------
// Implicit-GEMM 3x3 conv (stride 1, pad 1) via WMMA bf16.
//   Xp  : [B][64][64][CIN]  bf16 (channels innermost)
//   Wp  : [9][COUT][CIN]    bf16
//   out : f32, NCHW [B][COUT][64][64] region with fused epilogue
// Grid: (COUT/64, 32 row-pairs, B). Block 256 = 8 waves.
// Workgroup tile: 64 Cout x 128 pixels (2 rows). Each wave owns a 32x32 C tile
// (2 A-frags x 2 B-frags -> 4 WMMAs per 32-K step, 1.0 loads/WMMA).
//   waveM = wave&1  -> Cout subtile of 32
//   waveN = wave>>1 -> (row = y0 + (waveN>>1), xBase = (waveN&1)*32)
// Fragment layout (16-bit): lane&15 selects row(A)/col(B); lane>=16 takes K 16..31.
// MODE 0: gates (bias only). MODE 1: m_next = sigmoid(v)*tanh(c_next).
// MODE 2: decoded = lrelu(v); h_next = h_lstm*(1-a) + decoded*a + sr*h_orig.
// ---------------------------------------------------------------------------
template <int CIN, int COUT, int MODE>
__global__ __launch_bounds__(256) void conv3x3_wmma(
    const __bf16* __restrict__ Xp, const __bf16* __restrict__ Wp,
    const float* __restrict__ bias, float* __restrict__ out,
    const float* __restrict__ aux0, const float* __restrict__ aux1,
    const float* __restrict__ aux2, const float* __restrict__ aux3) {
    const int b      = blockIdx.z;
    const int coBase = blockIdx.x * 64;
    const int tid    = threadIdx.x;
    const int wave   = tid >> 5;
    const int lane   = tid & 31;
    const int waveN  = wave >> 1;
    const int mSub   = (wave & 1) * 32;
    const int y      = blockIdx.y * 2 + (waveN >> 1);
    const int xBase  = (waveN & 1) * 32;
    const int lr     = lane & 15;
    const int kOff   = (lane >> 4) * 16;   // K half select

    v8f acc00 = {};   // (m-tile 0, n-tile 0)
    v8f acc01 = {};
    v8f acc10 = {};
    v8f acc11 = {};
    bf16x16 zfrag = {};

    const int coRow0 = coBase + mSub + lr;        // A rows for this lane
    const int coRow1 = coRow0 + 16;
    const __bf16* xImg = Xp + (size_t)b * HH * WW * CIN;

    #pragma unroll
    for (int tap = 0; tap < 9; ++tap) {
        const int dy = tap / 3 - 1;
        const int dx = tap % 3 - 1;
        const int yy = y + dy;
        const bool rowOK = (yy >= 0) && (yy < HH);
        const __bf16* wRow0 = Wp + ((size_t)tap * COUT + coRow0) * CIN;
        const __bf16* wRow1 = Wp + ((size_t)tap * COUT + coRow1) * CIN;
        const int x0 = xBase + lr + dx;
        const int x1 = x0 + 16;
        const bool ok0 = rowOK && (x0 >= 0) && (x0 < WW);
        const bool ok1 = rowOK && (x1 >= 0) && (x1 < WW);
        const __bf16* xRow = xImg + (size_t)(rowOK ? yy : 0) * WW * CIN;
        // Pull the next tap's activation row toward the WGP while this tap's
        // WMMAs execute (lowers to global_prefetch_b8 on gfx1250).
        if (tap < 8) {
            int yn = y + (tap + 1) / 3 - 1;
            yn = yn < 0 ? 0 : (yn > HH - 1 ? HH - 1 : yn);
            const __bf16* pf = xImg + ((size_t)yn * WW + xBase + lr) * CIN;
            __builtin_prefetch((const void*)pf, 0, 3);
        }
        #pragma unroll
        for (int ci = 0; ci < CIN; ci += 32) {
            bf16x16 a0 = *(const bf16x16*)(wRow0 + ci + kOff);
            bf16x16 a1 = *(const bf16x16*)(wRow1 + ci + kOff);
            bf16x16 b0 = ok0 ? *(const bf16x16*)(xRow + (size_t)x0 * CIN + ci + kOff) : zfrag;
            bf16x16 b1 = ok1 ? *(const bf16x16*)(xRow + (size_t)x1 * CIN + ci + kOff) : zfrag;
            acc00 = wmma_bf16(a0, b0, acc00);
            acc01 = wmma_bf16(a0, b1, acc01);
            acc10 = wmma_bf16(a1, b0, acc10);
            acc11 = wmma_bf16(a1, b1, acc11);
        }
    }

    // C layout: lanes 0-15 -> rows r (VGPR r), N=lane; lanes 16-31 -> rows r+8.
    const int n    = lr;
    const int mOff = (lane >> 4) * 8;
    #pragma unroll
    for (int r = 0; r < 8; ++r) {
        #pragma unroll
        for (int tm = 0; tm < 2; ++tm) {
            const int co = coBase + mSub + tm * 16 + r + mOff;
            const float bval = bias[co];
            #pragma unroll
            for (int tn = 0; tn < 2; ++tn) {
                const int xo = xBase + tn * 16 + n;
                float v;
                if (tm == 0) v = (tn == 0 ? acc00[r] : acc01[r]);
                else         v = (tn == 0 ? acc10[r] : acc11[r]);
                v += bval;
                size_t oidx = (((size_t)b * COUT + co) * HH + y) * WW + xo;
                if (MODE == 0) {
                    out[oidx] = v;
                } else if (MODE == 1) {
                    float cn = aux0[oidx];                 // c_next, same NCHW index
                    out[oidx] = sigf(v) * tanhf(cn);
                } else {
                    float dec   = lrelu(v);
                    float alpha = sigf(aux2[0]);
                    float sr    = sigf(aux3[0]);
                    out[oidx] = aux0[oidx] * (1.0f - alpha) + dec * alpha + sr * aux1[oidx];
                }
            }
        }
    }
}

// ---------------------------------------------------------------------------
// ConvLSTM pointwise: c_next, h_lstm, and bf16 NHWC c_next for conv_m
// ---------------------------------------------------------------------------
__global__ void k_lstm_pointwise(const float* __restrict__ gates,
                                 const float* __restrict__ c,
                                 float* __restrict__ c_next_out,
                                 float* __restrict__ h_lstm,
                                 __bf16* __restrict__ c_next_bf) {
    int idx = blockIdx.x * blockDim.x + threadIdx.x;
    if (idx >= BB * IMG) return;
    int p  = idx % NPIX;
    int ch = (idx / NPIX) % HC;
    int b  = idx / IMG;
    size_t gbase = ((size_t)b * 4 * HC + ch) * NPIX + p;
    float ig = sigf(gates[gbase]);
    float fg = sigf(gates[gbase + (size_t)HC * NPIX]);
    float gg = tanhf(gates[gbase + (size_t)2 * HC * NPIX]);
    float og = sigf(gates[gbase + (size_t)3 * HC * NPIX]);
    float cn = fg * c[idx] + ig * gg;
    c_next_out[idx] = cn;
    h_lstm[idx] = og * tanhf(cn);
    c_next_bf[((size_t)b * NPIX + p) * HC + ch] = f2bf(cn);
}

// 4x4 average pool: h_lstm [B][HC][64][64] -> pooled [B][HC][16][16]
__global__ void k_pool(const float* __restrict__ h_lstm, float* __restrict__ pooled) {
    int idx = blockIdx.x * blockDim.x + threadIdx.x;
    if (idx >= BB * HC * 256) return;
    int px = idx % 16, py = (idx / 16) % 16;
    int ch = (idx / 256) % HC, b = idx / (256 * HC);
    const float* base = h_lstm + ((size_t)b * HC + ch) * NPIX;
    float s = 0.0f;
    #pragma unroll
    for (int iy = 0; iy < 4; ++iy)
        #pragma unroll
        for (int ix = 0; ix < 4; ++ix)
            s += base[(py * 4 + iy) * WW + (px * 4 + ix)];
    pooled[idx] = s * (1.0f / 16.0f);
}

// Encoder conv 3x3: pooled [B][64][16][16] -> lrelu -> e_bf16 [B][8192] (C*H*W order)
__global__ void k_enc_conv(const float* __restrict__ pooled,
                           const float* __restrict__ w, const float* __restrict__ bias,
                           __bf16* __restrict__ e_bf) {
    int idx = blockIdx.x * blockDim.x + threadIdx.x;
    if (idx >= BB * 32 * 256) return;
    int x = idx % 16, y = (idx / 16) % 16;
    int co = (idx / 256) % 32, b = idx / (256 * 32);
    float acc = bias[co];
    for (int ci = 0; ci < HC; ++ci) {
        const float* pin = pooled + ((size_t)b * HC + ci) * 256;
        const float* wr  = w + ((size_t)co * HC + ci) * 9;
        #pragma unroll
        for (int dy = -1; dy <= 1; ++dy) {
            int yy = y + dy;
            if (yy < 0 || yy > 15) continue;
            #pragma unroll
            for (int dx = -1; dx <= 1; ++dx) {
                int xx = x + dx;
                if (xx < 0 || xx > 15) continue;
                acc += wr[(dy + 1) * 3 + (dx + 1)] * pin[yy * 16 + xx];
            }
        }
    }
    e_bf[(size_t)b * 8192 + co * 256 + y * 16 + x] = f2bf(lrelu(acc));
}

// Encoder GEMM: [32 x 8192] @ [8192 x 64] with WMMA; one workgroup, 8 waves.
__global__ __launch_bounds__(256) void k_enc_gemm(const __bf16* __restrict__ e,
                                                  const __bf16* __restrict__ w,
                                                  const float* __restrict__ bias,
                                                  float* __restrict__ out) {
    const int tid  = threadIdx.x;
    const int wave = tid >> 5;
    const int lane = tid & 31;
    const int mT   = (wave & 1) * 16;   // batch tile
    const int nT   = (wave >> 1) * 16;  // SD tile
    const int lr   = lane & 15;
    const int kOff = (lane >> 4) * 16;
    v8f acc = {};
    const __bf16* aRow = e + (size_t)(mT + lr) * 8192;
    const __bf16* bRow = w + (size_t)(nT + lr) * 8192;
    for (int k = 0; k < 8192; k += 32) {
        bf16x16 a = *(const bf16x16*)(aRow + k + kOff);
        bf16x16 b = *(const bf16x16*)(bRow + k + kOff);
        acc = wmma_bf16(a, b, acc);
    }
    const int n = lr, mOff = (lane >> 4) * 8;
    #pragma unroll
    for (int r = 0; r < 8; ++r) {
        int bRowIdx = mT + r + mOff;
        int nn = nT + n;
        out[bRowIdx * SD + nn] = acc[r] + bias[nn];
    }
}

// LayerNorm over SD=64 per batch row
__global__ void k_enc_ln(const float* __restrict__ in, const float* __restrict__ g,
                         const float* __restrict__ bb, float* __restrict__ out) {
    __shared__ float s[SD];
    __shared__ float sMu, sVar;
    int b = blockIdx.x, t = threadIdx.x;
    s[t] = in[b * SD + t];
    __syncthreads();
    if (t == 0) {
        float mu = 0.0f;
        for (int j = 0; j < SD; ++j) mu += s[j];
        mu /= SD;
        float v = 0.0f;
        for (int j = 0; j < SD; ++j) { float d = s[j] - mu; v += d * d; }
        sMu = mu; sVar = v / SD;
    }
    __syncthreads();
    out[b * SD + t] = (s[t] - sMu) * rsqrtf(sVar + 1e-5f) * g[t] + bb[t];
}

// Fm = F_low @ F_high + I
__global__ void k_fm(const float* __restrict__ Fl, const float* __restrict__ Fh,
                     float* __restrict__ Fm) {
    int idx = blockIdx.x * blockDim.x + threadIdx.x;
    if (idx >= SD * SD) return;
    int i = idx / SD, j = idx % SD;
    float acc = (i == j) ? 1.0f : 0.0f;
    #pragma unroll 4
    for (int k = 0; k < 24; ++k) acc += Fl[i * 24 + k] * Fh[k * SD + j];
    Fm[idx] = acc;
}

// ---------------------------------------------------------------------------
// Per-batch EKF step: one workgroup per batch, everything in LDS (~50 KB).
// ---------------------------------------------------------------------------
__global__ __launch_bounds__(256) void k_kf_batch(
    const float* __restrict__ kf_state, const float* __restrict__ kf_cov,
    const float* __restrict__ Fm, const float* __restrict__ encoded,
    const float* __restrict__ nl1_w, const float* __restrict__ nl1_b,
    const float* __restrict__ nl_ln_g, const float* __restrict__ nl_ln_b,
    const float* __restrict__ nl2_w, const float* __restrict__ nl2_b,
    const float* __restrict__ nl_weight, const float* __restrict__ log_Q,
    const float* __restrict__ log_R, float* __restrict__ kfs_out,
    float* __restrict__ kfc_out) {
    __shared__ float sA[SD * SD];   // cov -> P_pred
    __shared__ float sB[SD * SD];   // T -> K
    __shared__ float sS[SD * SD];   // S -> Sinv -> U
    __shared__ float sVec[SD], sPred[SD], sInnov[SD], sRow[SD], sFac[SD], sRd[SD];
    __shared__ float sHid[128];
    __shared__ float sMu, sVar, sPivS;

    const int b = blockIdx.x, tid = threadIdx.x;

    if (tid < SD) { sVec[tid] = kf_state[b * SD + tid]; sRd[tid] = __expf(log_R[tid]); }
    __syncthreads();

    // hid = kf_state @ nl1_w.T + nl1_b
    if (tid < 128) {
        float acc = nl1_b[tid];
        for (int j = 0; j < SD; ++j) acc += nl1_w[tid * SD + j] * sVec[j];
        sHid[tid] = acc;
    }
    __syncthreads();
    if (tid == 0) {
        float mu = 0.0f;
        for (int j = 0; j < 128; ++j) mu += sHid[j];
        mu /= 128.0f;
        float v = 0.0f;
        for (int j = 0; j < 128; ++j) { float d = sHid[j] - mu; v += d * d; }
        sMu = mu; sVar = v / 128.0f;
    }
    __syncthreads();
    if (tid < 128) {
        float t = (sHid[tid] - sMu) * rsqrtf(sVar + 1e-5f) * nl_ln_g[tid] + nl_ln_b[tid];
        sHid[tid] = lrelu(t);
    }
    __syncthreads();

    // state_pred = kf_state @ Fm.T + sigmoid(nl_weight) * tanh(hid @ nl2_w.T + nl2_b)
    const float sigW = sigf(nl_weight[0]);
    if (tid < SD) {
        float acc = nl2_b[tid];
        for (int j = 0; j < 128; ++j) acc += nl2_w[tid * 128 + j] * sHid[j];
        float sp = sigW * tanhf(acc);
        for (int j = 0; j < SD; ++j) sp += sVec[j] * Fm[tid * SD + j];
        sPred[tid] = sp;
        sInnov[tid] = encoded[b * SD + tid] - sp;
    }
    for (int idx = tid; idx < SD * SD; idx += 256)
        sA[idx] = kf_cov[(size_t)b * SD * SD + idx];
    __syncthreads();

    // T = Fm @ cov
    for (int idx = tid; idx < SD * SD; idx += 256) {
        int i = idx >> 6, k = idx & 63;
        float acc = 0.0f;
        for (int j = 0; j < SD; ++j) acc += Fm[i * SD + j] * sA[j * SD + k];
        sB[idx] = acc;
    }
    __syncthreads();
    // P = T @ Fm^T + diag(Q)
    for (int idx = tid; idx < SD * SD; idx += 256) {
        int i = idx >> 6, l = idx & 63;
        float acc = 0.0f;
        for (int k = 0; k < SD; ++k) acc += sB[i * SD + k] * Fm[l * SD + k];
        if (i == l) acc += __expf(log_Q[i]);
        sA[idx] = acc;
    }
    __syncthreads();
    // S = P + diag(R) + 1e-5 I
    for (int idx = tid; idx < SD * SD; idx += 256) {
        int i = idx >> 6, l = idx & 63;
        float v = sA[idx];
        if (i == l) v += sRd[i] + 1e-5f;
        sS[idx] = v;
    }
    __syncthreads();
    // In-place Gauss-Jordan inversion of sS (regularized SPD; no pivoting)
    for (int k = 0; k < SD; ++k) {
        if (tid == 0) sPivS = 1.0f / sS[k * SD + k];
        __syncthreads();
        if (tid < SD) { sRow[tid] = sS[k * SD + tid] * sPivS; sFac[tid] = sS[tid * SD + k]; }
        __syncthreads();
        for (int idx = tid; idx < SD * SD; idx += 256) {
            int i = idx >> 6, j = idx & 63;
            float nv;
            if (i == k)      nv = (j == k) ? sPivS : sRow[j];
            else             nv = (j == k) ? (-sFac[i] * sPivS) : (sS[idx] - sFac[i] * sRow[j]);
            sS[idx] = nv;
        }
        __syncthreads();
    }
    // K = P @ Sinv
    for (int idx = tid; idx < SD * SD; idx += 256) {
        int i = idx >> 6, l = idx & 63;
        float acc = 0.0f;
        for (int j = 0; j < SD; ++j) acc += sA[i * SD + j] * sS[j * SD + l];
        sB[idx] = acc;
    }
    __syncthreads();
    // kf_state_next = state_pred + K @ innovation
    if (tid < SD) {
        float acc = sPred[tid];
        for (int j = 0; j < SD; ++j) acc += sB[tid * SD + j] * sInnov[j];
        kfs_out[b * SD + tid] = acc;
    }
    __syncthreads();
    // U = (I - K) @ P = P - K@P  -> sS (Sinv no longer needed)
    for (int idx = tid; idx < SD * SD; idx += 256) {
        int i = idx >> 6, k = idx & 63;
        float acc = sA[i * SD + k];
        for (int j = 0; j < SD; ++j) acc -= sB[i * SD + j] * sA[j * SD + k];
        sS[idx] = acc;
    }
    __syncthreads();
    // cov_next = U@(I-K)^T + K diag(R) K^T
    //          = U[i][l] - sum_k U[i][k] K[l][k] + sum_j K[i][j] Rd[j] K[l][j]
    for (int idx = tid; idx < SD * SD; idx += 256) {
        int i = idx >> 6, l = idx & 63;
        float acc = sS[i * SD + l];
        for (int k = 0; k < SD; ++k) acc -= sS[i * SD + k] * sB[l * SD + k];
        for (int j = 0; j < SD; ++j) acc += sB[i * SD + j] * sRd[j] * sB[l * SD + j];
        kfc_out[(size_t)b * SD * SD + idx] = acc;
    }
}

// Decoder linear: dlin[b][8192] = lrelu(kfs_next @ W.T + bias)
__global__ void k_dec_lin(const float* __restrict__ kfs, const float* __restrict__ w,
                          const float* __restrict__ bias, float* __restrict__ dlin) {
    int idx = blockIdx.x * blockDim.x + threadIdx.x;
    if (idx >= BB * 8192) return;
    int o = idx % 8192, b = idx / 8192;
    float acc = bias[o];
    const float* wr = w + (size_t)o * SD;
    const float* sr = kfs + b * SD;
    #pragma unroll 8
    for (int k = 0; k < SD; ++k) acc += wr[k] * sr[k];
    dlin[idx] = lrelu(acc);
}

// Bilinear 16x16 -> 64x64 (half-pixel, edge clamp), output bf16 NHWC [b][y][x][32]
__global__ void k_upsample(const float* __restrict__ dlin, __bf16* __restrict__ up) {
    int idx = blockIdx.x * blockDim.x + threadIdx.x;
    if (idx >= BB * NPIX * 32) return;
    int c = idx & 31;
    int pix = idx >> 5;
    int x = pix & 63, y = (pix >> 6) & 63, b = pix >> 12;
    float fy = (y + 0.5f) * 0.25f - 0.5f;
    float fx = (x + 0.5f) * 0.25f - 0.5f;
    int y0 = (int)floorf(fy), x0 = (int)floorf(fx);
    float wy = fy - (float)y0, wx = fx - (float)x0;
    int y0c = min(max(y0, 0), 15), y1c = min(max(y0 + 1, 0), 15);
    int x0c = min(max(x0, 0), 15), x1c = min(max(x0 + 1, 0), 15);
    const float* d = dlin + ((size_t)b * 32 + c) * 256;
    float v = (1.0f - wy) * ((1.0f - wx) * d[y0c * 16 + x0c] + wx * d[y0c * 16 + x1c]) +
              wy          * ((1.0f - wx) * d[y1c * 16 + x0c] + wx * d[y1c * 16 + x1c]);
    up[idx] = f2bf(v);
}

// ---------------------------------------------------------------------------
// Host launcher
// ---------------------------------------------------------------------------
extern "C" void kernel_launch(void* const* d_in, const int* in_sizes, int n_in,
                              void* d_out, int out_size, void* d_ws, size_t ws_size,
                              hipStream_t stream) {
    (void)in_sizes; (void)n_in; (void)out_size; (void)ws_size;
    const float* x          = (const float*)d_in[0];
    const float* h          = (const float*)d_in[1];
    const float* c          = (const float*)d_in[2];
    const float* m          = (const float*)d_in[3];
    const float* kf_state   = (const float*)d_in[4];
    const float* kf_cov     = (const float*)d_in[5];
    const float* w_in       = (const float*)d_in[6];
    const float* b_in       = (const float*)d_in[7];
    const float* w_gates    = (const float*)d_in[8];
    const float* b_gates    = (const float*)d_in[9];
    const float* w_m        = (const float*)d_in[10];
    const float* b_m        = (const float*)d_in[11];
    const float* enc_conv_w = (const float*)d_in[12];
    const float* enc_conv_b = (const float*)d_in[13];
    const float* enc_lin_w  = (const float*)d_in[14];
    const float* enc_lin_b  = (const float*)d_in[15];
    const float* enc_ln_g   = (const float*)d_in[16];
    const float* enc_ln_b   = (const float*)d_in[17];
    const float* dec_lin_w  = (const float*)d_in[18];
    const float* dec_lin_b  = (const float*)d_in[19];
    const float* dec_conv_w = (const float*)d_in[20];
    const float* dec_conv_b = (const float*)d_in[21];
    const float* F_low      = (const float*)d_in[22];
    const float* F_high     = (const float*)d_in[23];
    const float* nl1_w      = (const float*)d_in[24];
    const float* nl1_b      = (const float*)d_in[25];
    const float* nl_ln_g    = (const float*)d_in[26];
    const float* nl_ln_b    = (const float*)d_in[27];
    const float* nl2_w      = (const float*)d_in[28];
    const float* nl2_b      = (const float*)d_in[29];
    const float* nl_weight  = (const float*)d_in[30];
    const float* log_Q      = (const float*)d_in[31];
    const float* log_R      = (const float*)d_in[32];
    const float* kf_alpha   = (const float*)d_in[33];
    const float* res_weight = (const float*)d_in[34];

    // d_out layout: h_next | c_next | m_next | kf_state_next | kf_cov_next
    float* out      = (float*)d_out;
    float* o_h      = out;
    float* o_c      = out + (size_t)BB * IMG;
    float* o_m      = out + (size_t)2 * BB * IMG;
    float* o_kfs    = out + (size_t)3 * BB * IMG;
    float* o_kfc    = o_kfs + BB * SD;

    // Workspace carve-up (256B aligned)
    char* ws = (char*)d_ws;
    size_t off = 0;
    auto carve = [&](size_t bytes) {
        char* p = ws + off;
        off = (off + bytes + 255) & ~(size_t)255;
        return p;
    };
    __bf16* X192      = (__bf16*)carve((size_t)BB * NPIX * 192 * 2);
    __bf16* WpG       = (__bf16*)carve((size_t)9 * 256 * 192 * 2);
    __bf16* WpM       = (__bf16*)carve((size_t)9 * 64 * 64 * 2);
    __bf16* WpD       = (__bf16*)carve((size_t)9 * 64 * 32 * 2);
    __bf16* enclin_bf = (__bf16*)carve((size_t)SD * 8192 * 2);
    float*  gates     = (float*)carve((size_t)BB * 256 * NPIX * 4);
    float*  h_lstm    = (float*)carve((size_t)BB * IMG * 4);
    __bf16* cn_bf     = (__bf16*)carve((size_t)BB * NPIX * HC * 2);
    float*  pooled    = (float*)carve((size_t)BB * HC * 256 * 4);
    __bf16* e_bf      = (__bf16*)carve((size_t)BB * 8192 * 2);
    float*  enc_gemm  = (float*)carve((size_t)BB * SD * 4);
    float*  encoded   = (float*)carve((size_t)BB * SD * 4);
    float*  Fm        = (float*)carve((size_t)SD * SD * 4);
    float*  dlin      = (float*)carve((size_t)BB * 8192 * 4);
    __bf16* up_bf     = (__bf16*)carve((size_t)BB * NPIX * 32 * 2);

    const int T = 256;
    // 1. Pack input concat (with fused 1x1 input conv) to bf16 NHWC
    k_pack_x192<<<(BB * NPIX * 192 + T - 1) / T, T, 0, stream>>>(x, h, m, w_in, b_in, X192);
    // 2-4. Pack conv weights to [tap][Cout][Cin] bf16
    k_pack_taps<<<(256 * 192 * 9 + T - 1) / T, T, 0, stream>>>(w_gates, WpG, 256, 192);
    k_pack_taps<<<(64 * 64 * 9 + T - 1) / T, T, 0, stream>>>(w_m, WpM, 64, 64);
    k_pack_taps<<<(64 * 32 * 9 + T - 1) / T, T, 0, stream>>>(dec_conv_w, WpD, 64, 32);
    // 5. Encoder linear weight to bf16
    k_f32_to_bf16<<<(SD * 8192 + T - 1) / T, T, 0, stream>>>(enc_lin_w, enclin_bf, SD * 8192);
    // 6. conv_gates: the big WMMA conv (192 -> 256), 64x128 tile per workgroup
    conv3x3_wmma<192, 256, 0><<<dim3(4, HH / 2, BB), T, 0, stream>>>(
        X192, WpG, b_gates, gates, nullptr, nullptr, nullptr, nullptr);
    // 7. LSTM pointwise: c_next (to d_out), h_lstm, bf16 c_next
    k_lstm_pointwise<<<(BB * IMG + T - 1) / T, T, 0, stream>>>(gates, c, o_c, h_lstm, cn_bf);
    // 8. conv_m WMMA (64 -> 64) with fused m_next epilogue
    conv3x3_wmma<64, 64, 1><<<dim3(1, HH / 2, BB), T, 0, stream>>>(
        cn_bf, WpM, b_m, o_m, o_c, nullptr, nullptr, nullptr);
    // 9. 4x4 avg pool
    k_pool<<<(BB * HC * 256 + T - 1) / T, T, 0, stream>>>(h_lstm, pooled);
    // 10. Encoder conv (64 -> 32 on 16x16) + lrelu, write bf16 flat
    k_enc_conv<<<(BB * 32 * 256 + T - 1) / T, T, 0, stream>>>(pooled, enc_conv_w, enc_conv_b, e_bf);
    // 11. Encoder GEMM via WMMA (32 x 8192 x 64)
    k_enc_gemm<<<1, T, 0, stream>>>(e_bf, enclin_bf, enc_lin_b, enc_gemm);
    // 12. LayerNorm
    k_enc_ln<<<BB, SD, 0, stream>>>(enc_gemm, enc_ln_g, enc_ln_b, encoded);
    // 13. Fm = F_low @ F_high + I
    k_fm<<<(SD * SD + T - 1) / T, T, 0, stream>>>(F_low, F_high, Fm);
    // 14. EKF step per batch (state & cov to d_out)
    k_kf_batch<<<BB, T, 0, stream>>>(kf_state, kf_cov, Fm, encoded, nl1_w, nl1_b,
                                     nl_ln_g, nl_ln_b, nl2_w, nl2_b, nl_weight,
                                     log_Q, log_R, o_kfs, o_kfc);
    // 15. Decoder linear + lrelu
    k_dec_lin<<<(BB * 8192 + T - 1) / T, T, 0, stream>>>(o_kfs, dec_lin_w, dec_lin_b, dlin);
    // 16. Bilinear upsample to bf16 NHWC
    k_upsample<<<(BB * NPIX * 32 + T - 1) / T, T, 0, stream>>>(dlin, up_bf);
    // 17. Decoder conv WMMA (32 -> 64) with fused lrelu + h_next blend epilogue
    conv3x3_wmma<32, 64, 2><<<dim3(1, HH / 2, BB), T, 0, stream>>>(
        up_bf, WpD, dec_conv_b, o_h, h_lstm, h, kf_alpha, res_weight);
}
